// MemoryModule_56349970924091
// MI455X (gfx1250) — matrix operand order repaired and checked by reading, not compiled, not run on hardware
//
#include <hip/hip_runtime.h>
#include <hip/hip_bf16.h>
#include <math.h>

// ---------------------------------------------------------------------------
// MI455X (gfx1250) implementation: wave32 WMMA f16->f32 GEMM cores.
// Shapes: B=64, S=256, H=1024.
// ---------------------------------------------------------------------------

#define BB 64
#define SS 256
#define HH 1024
#define MTOT (BB * SS)      // 16384 rows of D flattened

typedef _Float16 half8  __attribute__((ext_vector_type(8)));
typedef _Float16 half16 __attribute__((ext_vector_type(16)));
typedef float    floatx8 __attribute__((ext_vector_type(8)));

union V16 { half16 v; half8 h[2]; };

// --------------------------- conversion kernels ----------------------------

__global__ void cvt_f32_to_f16_kernel(const float* __restrict__ src,
                                      _Float16* __restrict__ dst, size_t n) {
  size_t i = (size_t)blockIdx.x * blockDim.x + threadIdx.x;
  size_t stride = (size_t)gridDim.x * blockDim.x;
  for (; i < n; i += stride) dst[i] = (_Float16)src[i];
}

// src: K x N row-major f32  ->  dst: N x K row-major f16 (transposed)
__global__ void transpose_cvt_kernel(const float* __restrict__ src,
                                     _Float16* __restrict__ dst, int K, int N) {
  size_t total = (size_t)K * N;
  size_t stride = (size_t)gridDim.x * blockDim.x;
  for (size_t i = (size_t)blockIdx.x * blockDim.x + threadIdx.x; i < total; i += stride) {
    size_t k = i / N, n = i % N;
    dst[n * (size_t)K + k] = (_Float16)src[i];
  }
}

// --------------------------- fragment loaders ------------------------------
// A fragment (16x32 f16, row-major source, leading dim = ldk):
//   lane<16 : row M=lane,   j=0..7 -> K=k0+j,        j=8..15 -> K=k0+16+(j-8)
//   lane>=16: row M=lane-16,j=0..7 -> K=k0+8+j,      j=8..15 -> K=k0+24+(j-8)
// => two contiguous 8-half (16B) chunks at k0+8*(lane>>4) and k0+16+8*(lane>>4).
__device__ __forceinline__ half16 load_a_frag(const _Float16* __restrict__ A,
                                              int ldk, int row, int k0, int lanehi) {
  const _Float16* p = A + (size_t)row * ldk + k0 + 8 * lanehi;
  V16 f;
  f.h[0] = *(const half8*)(p);
  f.h[1] = *(const half8*)(p + 16);
  return f.v;
}

// B fragment (32x16 f16) from TRANSPOSED weight WT (N x K row-major):
//   lane: N = (lane&15)+col0 fixed, j=0..7 -> K = k0+8*(lane>>4)+j, j=8..15 -> K = k0+16+8*(lane>>4)+(j-8)
// => two contiguous 8-half (16B) chunks (same K order as A fragment).
__device__ __forceinline__ half16 load_b_frag(const _Float16* __restrict__ WT,
                                              int ldk, int col, int k0, int lanehi) {
  const _Float16* p = WT + (size_t)col * ldk + k0 + 8 * lanehi;
  V16 f;
  f.h[0] = *(const half8*)(p);
  f.h[1] = *(const half8*)(p + 16);
  return f.v;
}

// --------------------------- GEMM: Dr / Dw ---------------------------------
// out[M x N] = A(f16, MxK) @ WT^T + bias.  Wave = 16x64 tile (4 accums).
// grid: (N/64, Mtot/64), block: 128 (4 waves, wave w -> m-tile w).
__global__ void gemm16_bias_kernel(const _Float16* __restrict__ A,
                                   const _Float16* __restrict__ WT,
                                   const float* __restrict__ bias,
                                   float* __restrict__ out,
                                   int N, int K) {
  const int lane = threadIdx.x & 31, wid = threadIdx.x >> 5;
  const int lanelo = lane & 15, lanehi = lane >> 4;
  const int m0 = (blockIdx.y * 4 + wid) * 16;
  const int n0 = blockIdx.x * 64;
  const int arow = m0 + lanelo;

  floatx8 acc[4] = {};
  for (int k0 = 0; k0 < K; k0 += 32) {
    __builtin_prefetch(A + (size_t)arow * K + k0 + 256, 0, 0);
    half16 a = load_a_frag(A, K, arow, k0, lanehi);
#pragma unroll
    for (int c = 0; c < 4; ++c) {
      half16 b = load_b_frag(WT, K, n0 + c * 16 + lanelo, k0, lanehi);
      acc[c] = __builtin_amdgcn_wmma_f32_16x16x32_f16(
          false, a, false, b, (short)0, acc[c], false, false);
    }
  }
  // C layout: VGPR r, lane -> (M = r + 8*(lane>>4), N = lane&15)
  const int mrow = m0 + 8 * lanehi;
#pragma unroll
  for (int c = 0; c < 4; ++c) {
    const int ncol = n0 + c * 16 + lanelo;
    const float bv = bias[ncol];
#pragma unroll
    for (int r = 0; r < 8; ++r)
      out[(size_t)(mrow + r) * N + ncol] = acc[c][r] + bv;
  }
}

// --------------------------- gate scores Zs --------------------------------
// Zs[m] = tanh(z[m,:] @ W1 + b1) . W2 + b2, z built on the fly (K = 4H).
// Wave owns 16 rows, sweeps all 16 n-tiles sequentially, reduces with shfl.
__global__ void gate_kernel(const float* __restrict__ D, const float* __restrict__ Q,
                            const float* __restrict__ P,
                            const _Float16* __restrict__ W1T,
                            const float* __restrict__ b1, const float* __restrict__ W2,
                            const float* __restrict__ b2, float* __restrict__ Zs) {
  const int lane = threadIdx.x & 31, wid = threadIdx.x >> 5;
  const int lanelo = lane & 15, lanehi = lane >> 4;
  const int m0 = (blockIdx.x * 4 + wid) * 16;
  const int m = m0 + lanelo;
  const int b = m >> 8;  // S=256

  float rowsum[8] = {0.f, 0.f, 0.f, 0.f, 0.f, 0.f, 0.f, 0.f};

  for (int n0 = 0; n0 < HH; n0 += 64) {
    floatx8 acc[4] = {};
    for (int k0 = 0; k0 < 4 * HH; k0 += 32) {
      half16 a;
#pragma unroll
      for (int j = 0; j < 16; ++j) {
        const int k = k0 + j + 8 * ((j >> 3) + lanehi);
        const int seg = k >> 10, kk = k & (HH - 1);
        const float d = D[(size_t)m * HH + kk];
        const float q = Q[(size_t)b * HH + kk];
        const float p = P[(size_t)b * HH + kk];
        float v;
        if (seg == 0)      v = d * q;
        else if (seg == 1) v = d * p;
        else if (seg == 2) v = fabsf(d - q);
        else               v = fabsf(d - p);
        a[j] = (_Float16)v;
      }
#pragma unroll
      for (int c = 0; c < 4; ++c) {
        half16 bf = load_b_frag(W1T, 4 * HH, n0 + c * 16 + lanelo, k0, lanehi);
        acc[c] = __builtin_amdgcn_wmma_f32_16x16x32_f16(
            false, a, false, bf, (short)0, acc[c], false, false);
      }
    }
#pragma unroll
    for (int c = 0; c < 4; ++c) {
      const int ncol = n0 + c * 16 + lanelo;
      const float w2 = W2[ncol], vb1 = b1[ncol];
#pragma unroll
      for (int r = 0; r < 8; ++r)
        rowsum[r] += tanhf(acc[c][r] + vb1) * w2;
    }
  }
  // reduce over the 16 lanes of each half (N dimension)
#pragma unroll
  for (int r = 0; r < 8; ++r) {
    float v = rowsum[r];
    v += __shfl_xor(v, 8);
    v += __shfl_xor(v, 4);
    v += __shfl_xor(v, 2);
    v += __shfl_xor(v, 1);
    if (lanelo == 0) Zs[m0 + 8 * lanehi + r] = v + b2[0];
  }
}

// --------------------------- softmax over S --------------------------------
__global__ void softmax_kernel(const float* __restrict__ Zs, float* __restrict__ G) {
  __shared__ float sm[SS];
  const int b = blockIdx.x, t = threadIdx.x;
  const float v = Zs[b * SS + t];
  sm[t] = v;
  __syncthreads();
  for (int s = SS / 2; s > 0; s >>= 1) {
    if (t < s) sm[t] = fmaxf(sm[t], sm[t + s]);
    __syncthreads();
  }
  const float mx = sm[0];
  __syncthreads();
  const float e = expf(v - mx);
  sm[t] = e;
  __syncthreads();
  for (int s = SS / 2; s > 0; s >>= 1) {
    if (t < s) sm[t] += sm[t + s];
    __syncthreads();
  }
  G[b * SS + t] = e / sm[0];
}

// --------------------------- GRU-style scan --------------------------------
// Block = 16 batch rows (one WMMA M-tile), all N=1024 columns, h in LDS (f16).
// 32 waves x 32 columns each; 256 serial steps with workgroup barriers.
__global__ __launch_bounds__(1024) void scan_kernel(
    const float* __restrict__ Dr, const float* __restrict__ Dw,
    const float* __restrict__ G,
    const _Float16* __restrict__ UrT, const _Float16* __restrict__ UT,
    const float* __restrict__ bur, const float* __restrict__ bu,
    float* __restrict__ hs_out, float* __restrict__ Cout) {
  __shared__ __align__(32) _Float16 hsh[16 * HH];  // 32 KB

  const int tid = threadIdx.x;
  const int lane = tid & 31, wid = tid >> 5;
  const int lanelo = lane & 15, lanehi = lane >> 4;
  const int b0 = blockIdx.x * 16;
  const int n0 = wid * 32;

  for (int i = tid; i < 16 * HH; i += blockDim.x) hsh[i] = (_Float16)0.f;
  __syncthreads();

  for (int t = 0; t < SS; ++t) {
    floatx8 accR[2] = {};
    floatx8 accU[2] = {};
    const _Float16* arow = hsh + (size_t)lanelo * HH;
    for (int k0 = 0; k0 < HH; k0 += 32) {
      V16 af;
      const _Float16* ap = arow + k0 + 8 * lanehi;
      af.h[0] = *(const half8*)(ap);
      af.h[1] = *(const half8*)(ap + 16);
#pragma unroll
      for (int c = 0; c < 2; ++c) {
        const int col = n0 + c * 16 + lanelo;
        half16 bR = load_b_frag(UrT, HH, col, k0, lanehi);
        half16 bU = load_b_frag(UT, HH, col, k0, lanehi);
        accR[c] = __builtin_amdgcn_wmma_f32_16x16x32_f16(
            false, af.v, false, bR, (short)0, accR[c], false, false);
        accU[c] = __builtin_amdgcn_wmma_f32_16x16x32_f16(
            false, af.v, false, bU, (short)0, accU[c], false, false);
      }
    }

    float hnew[2][8];
#pragma unroll
    for (int c = 0; c < 2; ++c) {
      const int ncol = n0 + c * 16 + lanelo;
      const float vbur = bur[ncol], vbu = bu[ncol];
#pragma unroll
      for (int r = 0; r < 8; ++r) {
        const int m = r + 8 * lanehi;
        const int bb = b0 + m;
        const size_t idx = ((size_t)bb * SS + t) * HH + ncol;
        const float dr = Dr[idx], dw = Dw[idx];
        const float g = G[bb * SS + t];
        const float hold = (float)hsh[(size_t)m * HH + ncol];
        const float rg = 1.f / (1.f + expf(-(dr + accR[c][r] + vbur)));
        const float ht = tanhf(dw + rg * (accU[c][r] + vbu));
        const float hn = g * ht + (1.f - g) * hold;
        hs_out[idx] = hn;
        hnew[c][r] = hn;
      }
    }
    __syncthreads();  // all waves done reading h for this step
#pragma unroll
    for (int c = 0; c < 2; ++c) {
      const int ncol = n0 + c * 16 + lanelo;
#pragma unroll
      for (int r = 0; r < 8; ++r)
        hsh[(size_t)(r + 8 * lanehi) * HH + ncol] = (_Float16)hnew[c][r];
    }
    __syncthreads();  // new h visible to all waves
  }

  for (int i = tid; i < 16 * HH; i += blockDim.x) {
    const int m = i >> 10, n = i & (HH - 1);
    Cout[(size_t)(b0 + m) * HH + n] = (float)hsh[i];
  }
}

// --------------------------- final projection ------------------------------
// next_mem = relu([P, C, Q] @ Wm + bm)   (M=64, K=3072, N=1024)
__global__ void final_kernel(const float* __restrict__ P, const float* __restrict__ C,
                             const float* __restrict__ Q,
                             const _Float16* __restrict__ WmT,
                             const float* __restrict__ bm, float* __restrict__ out) {
  const int lane = threadIdx.x & 31, wid = threadIdx.x >> 5;
  const int lanelo = lane & 15, lanehi = lane >> 4;
  const int m0 = wid * 16;
  const int n0 = blockIdx.x * 64;
  const int m = m0 + lanelo;

  floatx8 acc[4] = {};
  for (int k0 = 0; k0 < 3 * HH; k0 += 32) {
    half16 a;
#pragma unroll
    for (int j = 0; j < 16; ++j) {
      const int k = k0 + j + 8 * ((j >> 3) + lanehi);
      const int seg = k >> 10, kk = k & (HH - 1);
      const float* src = (seg == 0) ? P : (seg == 1) ? C : Q;
      a[j] = (_Float16)src[(size_t)m * HH + kk];
    }
#pragma unroll
    for (int c = 0; c < 4; ++c) {
      half16 b = load_b_frag(WmT, 3 * HH, n0 + c * 16 + lanelo, k0, lanehi);
      acc[c] = __builtin_amdgcn_wmma_f32_16x16x32_f16(
          false, a, false, b, (short)0, acc[c], false, false);
    }
  }
  const int mrow = m0 + 8 * lanehi;
#pragma unroll
  for (int c = 0; c < 4; ++c) {
    const int ncol = n0 + c * 16 + lanelo;
    const float bv = bm[ncol];
#pragma unroll
    for (int r = 0; r < 8; ++r)
      out[(size_t)(mrow + r) * HH + ncol] = fmaxf(acc[c][r] + bv, 0.f);
  }
}

// --------------------------- host-side launcher ----------------------------

extern "C" void kernel_launch(void* const* d_in, const int* in_sizes, int n_in,
                              void* d_out, int out_size, void* d_ws, size_t ws_size,
                              hipStream_t stream) {
  (void)in_sizes; (void)n_in; (void)out_size; (void)ws_size;

  const float* D  = (const float*)d_in[0];
  const float* Q  = (const float*)d_in[1];
  const float* P  = (const float*)d_in[2];   // prev_mem
  const float* Wr = (const float*)d_in[3];
  const float* br = (const float*)d_in[4];
  const float* Ur = (const float*)d_in[5];
  const float* bur= (const float*)d_in[6];
  const float* Wx = (const float*)d_in[7];
  const float* bx = (const float*)d_in[8];
  const float* U  = (const float*)d_in[9];
  const float* bu = (const float*)d_in[10];
  const float* W1 = (const float*)d_in[11];
  const float* b1 = (const float*)d_in[12];
  const float* W2 = (const float*)d_in[13];
  const float* b2 = (const float*)d_in[14];
  const float* Wm = (const float*)d_in[15];
  const float* bm = (const float*)d_in[16];

  float* out_next = (float*)d_out;                 // (B,1,H)   = 65536 f32
  float* out_hs   = (float*)d_out + (size_t)BB * HH;  // (B,S,H)

  // workspace carve (256B-aligned slices)
  char* w = (char*)d_ws;
  auto carve = [&](size_t bytes) -> void* {
    void* p = (void*)w;
    w += (bytes + 255) & ~(size_t)255;
    return p;
  };
  _Float16* Dh  = (_Float16*)carve((size_t)MTOT * HH * 2);
  _Float16* WrT = (_Float16*)carve((size_t)HH * HH * 2);
  _Float16* WxT = (_Float16*)carve((size_t)HH * HH * 2);
  _Float16* UrT = (_Float16*)carve((size_t)HH * HH * 2);
  _Float16* UT  = (_Float16*)carve((size_t)HH * HH * 2);
  _Float16* W1T = (_Float16*)carve((size_t)4 * HH * HH * 2);
  _Float16* WmT = (_Float16*)carve((size_t)3 * HH * HH * 2);
  float* Dr  = (float*)carve((size_t)MTOT * HH * 4);
  float* Dw  = (float*)carve((size_t)MTOT * HH * 4);
  float* Zs  = (float*)carve((size_t)MTOT * 4);
  float* G   = (float*)carve((size_t)MTOT * 4);
  float* Cb  = (float*)carve((size_t)BB * HH * 4);

  // 1) precision staging: D -> f16; weights -> transposed f16 (N x K)
  cvt_f32_to_f16_kernel<<<4096, 256, 0, stream>>>(D, Dh, (size_t)MTOT * HH);
  transpose_cvt_kernel<<<1024, 256, 0, stream>>>(Wr, WrT, HH, HH);
  transpose_cvt_kernel<<<1024, 256, 0, stream>>>(Wx, WxT, HH, HH);
  transpose_cvt_kernel<<<1024, 256, 0, stream>>>(Ur, UrT, HH, HH);
  transpose_cvt_kernel<<<1024, 256, 0, stream>>>(U,  UT,  HH, HH);
  transpose_cvt_kernel<<<2048, 256, 0, stream>>>(W1, W1T, 4 * HH, HH);
  transpose_cvt_kernel<<<2048, 256, 0, stream>>>(Wm, WmT, 3 * HH, HH);

  // 2) Dr = D@Wr + br ; Dw = D@Wx + bx   (WMMA)
  dim3 ggrid(HH / 64, MTOT / 64);
  gemm16_bias_kernel<<<ggrid, 128, 0, stream>>>(Dh, WrT, br, Dr, HH, HH);
  gemm16_bias_kernel<<<ggrid, 128, 0, stream>>>(Dh, WxT, bx, Dw, HH, HH);

  // 3) gate scores + softmax
  gate_kernel<<<MTOT / 64, 128, 0, stream>>>(D, Q, P, W1T, b1, W2, b2, Zs);
  softmax_kernel<<<BB, SS, 0, stream>>>(Zs, G);

  // 4) sequential scan: 4 workgroups x 16 batch rows, h resident in LDS
  scan_kernel<<<BB / 16, 1024, 0, stream>>>(Dr, Dw, G, UrT, UT, bur, bu,
                                            out_hs, Cb);

  // 5) next_mem = relu([P, C, Q] @ Wm + bm)
  final_kernel<<<HH / 64, 128, 0, stream>>>(P, Cb, Q, WmT, bm, out_next);
}
